// MaskedMultiHeadAttention_59270548685592
// MI455X (gfx1250) — matrix operand order repaired and checked by reading, not compiled, not run on hardware
//
#include <hip/hip_runtime.h>

// Problem constants (from reference): S=1024, P=1024, B=2, E=1024, H=16, I=64, J=2048
#define Sc  1024
#define Pc  1024
#define Bc  2
#define Ec  1024
#define Hc  16
#define Ic  64
#define Jc  2048
#define HIc 1024   // H*I
#define SCALEc 0.125f
#define EPSc 1e-5f

typedef _Float16 half_t;
typedef __attribute__((ext_vector_type(16))) _Float16 v16h;
typedef __attribute__((ext_vector_type(8)))  _Float16 v8h;
typedef __attribute__((ext_vector_type(8)))  float    v8f;

// ---------------------------------------------------------------------------
// CDNA5 WMMA helpers (wave32, 16x16x32 f16 -> f32)
// ---------------------------------------------------------------------------
__device__ __forceinline__ v8f wmma_f16(v16h a, v16h b, v8f c) {
  return __builtin_amdgcn_wmma_f32_16x16x32_f16(
      /*neg_a=*/false, a, /*neg_b=*/false, b,
      /*c_mod=*/(short)0, c, /*reuse_a=*/false, /*reuse_b=*/false);
}

// A-matrix (16x32 f16) from row-major memory, lda in elements.
// Layout: lane L -> row M = L&15; half h -> K = (L>>4)*16 + h
__device__ __forceinline__ v16h load_a_rm(const half_t* A, int lda) {
  const int lane = threadIdx.x & 31;
  const half_t* p = A + (size_t)(lane & 15) * lda + ((lane >> 4) << 4);
  v8h lo = *(const v8h*)p;          // K = kbase .. kbase+7
  v8h hi = *(const v8h*)(p + 8);    // K = kbase+8 .. kbase+15
  v16h r;
#pragma unroll
  for (int e = 0; e < 8; ++e) { r[e] = lo[e]; r[e + 8] = hi[e]; }
  return r;
}

// B-matrix (32x16) where B[k][n] = W[n*ldw + k]  (W row-major (N,K), i.e. GEMM with W^T).
// Layout: lane L -> col N = L&15; half h -> K = (L>>4)*16 + h  (one contiguous 32B run)
__device__ __forceinline__ v16h load_b_wT(const half_t* W, int ldw) {
  const int lane = threadIdx.x & 31;
  return *(const v16h*)(W + (size_t)(lane & 15) * ldw + ((lane >> 4) << 4));
}

// B-matrix (32x16) where B[k][n] = Bm[k*ldb + n]  (Bm row-major (K,N)) — strided gather
__device__ __forceinline__ v16h load_b_rm(const half_t* Bm, int ldb) {
  const int lane = threadIdx.x & 31;
  const half_t* p = Bm + (size_t)((lane >> 4) << 4) * ldb + (lane & 15);
  v16h r;
#pragma unroll
  for (int h = 0; h < 16; ++h) r[h] = p[(size_t)h * ldb];
  return r;
}

// ---------------------------------------------------------------------------
// 32x64 output tile per wave: 2 A-subtiles x 4 B-subtiles, 8 accumulators.
// Per 32-wide k-step: 4 A b128 loads + 8 B b128 loads feed 8 WMMAs.
// acc[mi][ni] holds rows m0+mi*16+.., cols n0+ni*16+..
// ---------------------------------------------------------------------------
__device__ __forceinline__ void gemm_awt_32x64(const half_t* A, int lda,
                                               const half_t* W, int ldw, int K,
                                               v8f acc[2][4]) {
#pragma unroll 4
  for (int k = 0; k < K; k += 32) {
    // prefetch next k-step into WGP$ (global_prefetch_b8; no counter cost)
    if (k + 32 < K) {
      __builtin_prefetch(A + k + 32, 0, 3);
      __builtin_prefetch(A + (size_t)16 * lda + k + 32, 0, 3);
      __builtin_prefetch(W + k + 32, 0, 3);
    }
    v16h a0 = load_a_rm(A + k, lda);
    v16h a1 = load_a_rm(A + (size_t)16 * lda + k, lda);
#pragma unroll
    for (int n = 0; n < 4; ++n) {
      v16h b = load_b_wT(W + (size_t)(n * 16) * ldw + k, ldw);
      acc[0][n] = wmma_f16(a0, b, acc[0][n]);
      acc[1][n] = wmma_f16(a1, b, acc[1][n]);
    }
  }
}

// C/D layout: element r of a v8f lives at (M = r + 8*(lane>>4), N = lane&15)

// ---------------------------------------------------------------------------
// Prep kernels
// ---------------------------------------------------------------------------
__global__ void k_cvt(const float* __restrict__ in, half_t* __restrict__ out, int n) {
  int i = blockIdx.x * blockDim.x + threadIdx.x;
  if (i < n) out[i] = (half_t)in[i];
}

// iw = concat(memory (P,B,E), x (S,B,E)) along axis 0, to f16
__global__ void k_build_iw(const float* __restrict__ mem, const float* __restrict__ x,
                           half_t* __restrict__ iw, int n) {
  int i = blockIdx.x * blockDim.x + threadIdx.x;
  if (i >= n) return;
  const int pbe = Pc * Bc * Ec;
  iw[i] = (half_t)(i < pbe ? mem[i] : x[i - pbe]);
}

// ---------------------------------------------------------------------------
// GEMM kernels (one wave = one 32x64 output tile)
// ---------------------------------------------------------------------------
// kv = iw(J*B, E) @ Wkv^T(E, 2HI); split/scatter to k,v in (B,H,J,I) layout
__global__ __launch_bounds__(32) void k_gemm_kv(const half_t* __restrict__ iw,
                                                const half_t* __restrict__ Wkv,
                                                half_t* __restrict__ kh,
                                                half_t* __restrict__ vh) {
  const int n0 = blockIdx.x * 64, m0 = blockIdx.y * 32;
  v8f acc[2][4] = {};
  gemm_awt_32x64(iw + (size_t)m0 * Ec, Ec, Wkv + (size_t)n0 * Ec, Ec, Ec, acc);
  const int lane = threadIdx.x & 31;
  const int cn = lane & 15, coff = (lane >> 4) << 3;
#pragma unroll
  for (int mi = 0; mi < 2; ++mi)
#pragma unroll
    for (int ni = 0; ni < 4; ++ni)
#pragma unroll
      for (int r = 0; r < 8; ++r) {
        const int row = m0 + mi * 16 + r + coff;   // row = j*B + b
        const int col = n0 + ni * 16 + cn;         // col in [0, 2*HI)
        const int b = row % Bc, j = row / Bc;
        const int sel = col / HIc, hi = col % HIc;
        const int h = hi / Ic, i = hi % Ic;
        half_t* dst = sel ? vh : kh;
        dst[(((size_t)(b * Hc + h)) * Jc + j) * Ic + i] = (half_t)acc[mi][ni][r];
      }
}

// q = x(S*B, E) @ Wq^T; write qu = q+u and qv = q+v  (both (S,B,H*I) row-major)
__global__ __launch_bounds__(32) void k_gemm_q(const half_t* __restrict__ xh,
                                               const half_t* __restrict__ Wq,
                                               const float* __restrict__ u,
                                               const float* __restrict__ v,
                                               half_t* __restrict__ qu,
                                               half_t* __restrict__ qv) {
  const int n0 = blockIdx.x * 64, m0 = blockIdx.y * 32;
  v8f acc[2][4] = {};
  gemm_awt_32x64(xh + (size_t)m0 * Ec, Ec, Wq + (size_t)n0 * Ec, Ec, Ec, acc);
  const int lane = threadIdx.x & 31;
  const int cn = lane & 15, coff = (lane >> 4) << 3;
#pragma unroll
  for (int mi = 0; mi < 2; ++mi)
#pragma unroll
    for (int ni = 0; ni < 4; ++ni)
#pragma unroll
      for (int r = 0; r < 8; ++r) {
        const int row = m0 + mi * 16 + r + coff;
        const int col = n0 + ni * 16 + cn;
        const float a = acc[mi][ni][r];
        qu[(size_t)row * HIc + col] = (half_t)(a + u[col]);
        qv[(size_t)row * HIc + col] = (half_t)(a + v[col]);
      }
}

// pos = rel_pos(J, E) @ Wp^T; scatter to (H, J, I)
__global__ __launch_bounds__(32) void k_gemm_pos(const half_t* __restrict__ rp,
                                                 const half_t* __restrict__ Wp,
                                                 half_t* __restrict__ posh) {
  const int n0 = blockIdx.x * 64, m0 = blockIdx.y * 32;
  v8f acc[2][4] = {};
  gemm_awt_32x64(rp + (size_t)m0 * Ec, Ec, Wp + (size_t)n0 * Ec, Ec, Ec, acc);
  const int lane = threadIdx.x & 31;
  const int cn = lane & 15, coff = (lane >> 4) << 3;
#pragma unroll
  for (int mi = 0; mi < 2; ++mi)
#pragma unroll
    for (int ni = 0; ni < 4; ++ni)
#pragma unroll
      for (int r = 0; r < 8; ++r) {
        const int j = m0 + mi * 16 + r + coff;
        const int col = n0 + ni * 16 + cn;
        const int h = col / Ic, i = col % Ic;
        posh[((size_t)h * Jc + j) * Ic + i] = (half_t)acc[mi][ni][r];
      }
}

// D[b,h,s,jp] = sum_i qv[s,b,h,i] * pos[h,jp,i]   (unshifted positional scores)
__global__ __launch_bounds__(32) void k_gemm_D(const half_t* __restrict__ qv,
                                               const half_t* __restrict__ posh,
                                               half_t* __restrict__ Dh) {
  const int n0 = blockIdx.x * 64, m0 = blockIdx.y * 32;
  const int bh = blockIdx.z;
  const int b = bh / Hc, h = bh % Hc;
  const half_t* A = qv + (((size_t)m0 * Bc + b) * Hc + h) * Ic;  // row stride B*H*I
  v8f acc[2][4] = {};
  gemm_awt_32x64(A, Bc * Hc * Ic, posh + ((size_t)h * Jc + n0) * Ic, Ic, Ic, acc);
  const int lane = threadIdx.x & 31;
  const int cn = lane & 15, coff = (lane >> 4) << 3;
#pragma unroll
  for (int mi = 0; mi < 2; ++mi)
#pragma unroll
    for (int ni = 0; ni < 4; ++ni)
#pragma unroll
      for (int r = 0; r < 8; ++r)
        Dh[((size_t)bh * Sc + (m0 + mi * 16 + r + coff)) * Jc + (n0 + ni * 16 + cn)] =
            (half_t)acc[mi][ni][r];
}

// out = x + ctx(S*B, HI) @ Wo^T(HI, E)   (f32, pre-LayerNorm)
__global__ __launch_bounds__(32) void k_gemm_out(const half_t* __restrict__ ctx,
                                                 const half_t* __restrict__ Wo,
                                                 const float* __restrict__ x,
                                                 float* __restrict__ outf) {
  const int n0 = blockIdx.x * 64, m0 = blockIdx.y * 32;
  v8f acc[2][4] = {};
  gemm_awt_32x64(ctx + (size_t)m0 * HIc, HIc, Wo + (size_t)n0 * HIc, HIc, HIc, acc);
  const int lane = threadIdx.x & 31;
  const int cn = lane & 15, coff = (lane >> 4) << 3;
#pragma unroll
  for (int mi = 0; mi < 2; ++mi)
#pragma unroll
    for (int ni = 0; ni < 4; ++ni)
#pragma unroll
      for (int r = 0; r < 8; ++r) {
        const int row = m0 + mi * 16 + r + coff;
        const int col = n0 + ni * 16 + cn;
        outf[(size_t)row * Ec + col] = acc[mi][ni][r] + x[(size_t)row * Ec + col];
      }
}

// ---------------------------------------------------------------------------
// Fused flash-style attention per (b, h, 16-query tile).
// attn[s,j] = (content[s,j] + D[s, j+S-1-s]) * SCALE  for j <= s+P, else -inf
// Online softmax; ctx accumulated in f32; one wave per block.
// ---------------------------------------------------------------------------
__global__ __launch_bounds__(32) void k_attn(const half_t* __restrict__ qu,
                                             const half_t* __restrict__ kmat,
                                             const half_t* __restrict__ vmat,
                                             const half_t* __restrict__ Dmat,
                                             half_t* __restrict__ ctx) {
  __shared__ __align__(16) half_t Plds[16 * 32];
  const int s0 = blockIdx.x * 16;
  const int bh = blockIdx.y;
  const int b = bh / Hc, h = bh % Hc;
  const int lane = threadIdx.x & 31;
  const int cn = lane & 15;
  const int coff = (lane >> 4) << 3;

  const half_t* Kb = kmat + (size_t)bh * Jc * Ic;   // (J, I) for this (b,h)
  const half_t* Vb = vmat + (size_t)bh * Jc * Ic;
  const half_t* Db = Dmat + (size_t)bh * Sc * Jc;

  // Query tile (q+u) as two A-layout chunks (K = i in [0,32) and [32,64))
  const half_t* Arow = qu + (((size_t)s0 * Bc + b) * Hc + h) * Ic;
  const int lda = Bc * Hc * Ic;
  v16h a0 = load_a_rm(Arow, lda);
  v16h a1 = load_a_rm(Arow + 32, lda);

  float m[8], l[8];
  v8f o0 = {}, o1 = {}, o2 = {}, o3 = {};
#pragma unroll
  for (int r = 0; r < 8; ++r) { m[r] = -__builtin_inff(); l[r] = 0.f; }

  const int jend = s0 + 16 + Pc;        // max valid key index + 1 for this tile
  const int ntiles = (jend + 31) >> 5;  // 32 keys per iteration
  for (int t = 0; t < ntiles; ++t) {
    const int j0 = t << 5;
    // content scores: Qu(16x64) @ K^T -> two 16x16 chunks (key cols 0..15, 16..31)
    v8f sc0 = {}, sc1 = {};
    sc0 = wmma_f16(a0, load_b_wT(Kb + (size_t)j0 * Ic, Ic), sc0);
    sc0 = wmma_f16(a1, load_b_wT(Kb + (size_t)j0 * Ic + 32, Ic), sc0);
    sc1 = wmma_f16(a0, load_b_wT(Kb + (size_t)(j0 + 16) * Ic, Ic), sc1);
    sc1 = wmma_f16(a1, load_b_wT(Kb + (size_t)(j0 + 16) * Ic + 32, Ic), sc1);

    float p0[8], p1[8];
#pragma unroll
    for (int r = 0; r < 8; ++r) {
      const int s = s0 + r + coff;
      const int jA = j0 + cn;
      const int jB = j0 + 16 + cn;
      float v0 = -__builtin_inff(), v1 = -__builtin_inff();
      if (jA <= s + Pc)   // rel-shift gather: jp = j + (S-1) - s
        v0 = (sc0[r] + (float)Db[(size_t)s * Jc + (jA + (Sc - 1) - s)]) * SCALEc;
      if (jB <= s + Pc)
        v1 = (sc1[r] + (float)Db[(size_t)s * Jc + (jB + (Sc - 1) - s)]) * SCALEc;
      // row max across the 16 lanes of this half-wave
      float mx = fmaxf(v0, v1);
#pragma unroll
      for (int off = 1; off < 16; off <<= 1)
        mx = fmaxf(mx, __shfl_xor(mx, off, 32));
      const float nm = fmaxf(m[r], mx);
      const float al = __expf(m[r] - nm);
      const float e0 = __expf(v0 - nm);
      const float e1 = __expf(v1 - nm);
      float rs = e0 + e1;
#pragma unroll
      for (int off = 1; off < 16; off <<= 1)
        rs += __shfl_xor(rs, off, 32);
      l[r] = l[r] * al + rs;
      m[r] = nm;
      o0[r] *= al; o1[r] *= al; o2[r] *= al; o3[r] *= al;
      p0[r] = e0; p1[r] = e1;
    }

    // transpose P (C-layout regs) -> A-layout via LDS, then P(16x32) @ V(32x64)
#pragma unroll
    for (int r = 0; r < 8; ++r) {
      Plds[(r + coff) * 32 + cn]      = (half_t)p0[r];
      Plds[(r + coff) * 32 + 16 + cn] = (half_t)p1[r];
    }
    __asm__ volatile("s_wait_dscnt 0" ::: "memory");  // CDNA5 split DS counter wait
    v16h pa = load_a_rm(Plds, 32);
    o0 = wmma_f16(pa, load_b_rm(Vb + (size_t)j0 * Ic + 0,  Ic), o0);
    o1 = wmma_f16(pa, load_b_rm(Vb + (size_t)j0 * Ic + 16, Ic), o1);
    o2 = wmma_f16(pa, load_b_rm(Vb + (size_t)j0 * Ic + 32, Ic), o2);
    o3 = wmma_f16(pa, load_b_rm(Vb + (size_t)j0 * Ic + 48, Ic), o3);
  }

  // normalize and store ctx (S, B, H*I)
#pragma unroll
  for (int r = 0; r < 8; ++r) {
    const int s = s0 + r + coff;
    const float inv = 1.0f / l[r];
    half_t* dst = ctx + ((size_t)s * Bc + b) * HIc + (size_t)h * Ic + cn;
    dst[0]  = (half_t)(o0[r] * inv);
    dst[16] = (half_t)(o1[r] * inv);
    dst[32] = (half_t)(o2[r] * inv);
    dst[48] = (half_t)(o3[r] * inv);
  }
}

// ---------------------------------------------------------------------------
// LayerNorm over E per (s,b) row
// ---------------------------------------------------------------------------
__global__ __launch_bounds__(256) void k_layernorm(const float* __restrict__ in,
                                                   const float* __restrict__ gamma,
                                                   const float* __restrict__ beta,
                                                   float* __restrict__ out) {
  __shared__ float red[256];
  const int row = blockIdx.x, tid = threadIdx.x;
  const float* p = in + (size_t)row * Ec;
  float s = 0.f;
  for (int i = tid; i < Ec; i += 256) s += p[i];
  red[tid] = s; __syncthreads();
  for (int w = 128; w > 0; w >>= 1) { if (tid < w) red[tid] += red[tid + w]; __syncthreads(); }
  const float mu = red[0] / Ec;
  __syncthreads();
  float vs = 0.f;
  for (int i = tid; i < Ec; i += 256) { float d = p[i] - mu; vs += d * d; }
  red[tid] = vs; __syncthreads();
  for (int w = 128; w > 0; w >>= 1) { if (tid < w) red[tid] += red[tid + w]; __syncthreads(); }
  const float rstd = rsqrtf(red[0] / Ec + EPSc);
  for (int i = tid; i < Ec; i += 256)
    out[(size_t)row * Ec + i] = (p[i] - mu) * rstd * gamma[i] + beta[i];
}

// ---------------------------------------------------------------------------
// Host launcher
// ---------------------------------------------------------------------------
extern "C" void kernel_launch(void* const* d_in, const int* in_sizes, int n_in,
                              void* d_out, int out_size, void* d_ws, size_t ws_size,
                              hipStream_t stream) {
  (void)in_sizes; (void)n_in; (void)out_size; (void)ws_size;
  const float* x       = (const float*)d_in[0];
  const float* rel_pos = (const float*)d_in[1];
  const float* memory  = (const float*)d_in[2];
  const float* u       = (const float*)d_in[3];
  const float* v       = (const float*)d_in[4];
  // d_in[5] = mask (bool) — recomputed analytically (j > s+P), not read
  const float* Wkv     = (const float*)d_in[6];
  const float* Wq      = (const float*)d_in[7];
  const float* Wp      = (const float*)d_in[8];
  const float* Wo      = (const float*)d_in[9];
  const float* gamma   = (const float*)d_in[10];
  const float* beta    = (const float*)d_in[11];

  char* wsb = (char*)d_ws;
  size_t off = 0;
  auto allocB = [&](size_t bytes) -> void* {
    void* p = wsb + off;
    off += (bytes + 255) & ~(size_t)255;
    return p;
  };
  half_t* xh   = (half_t*)allocB((size_t)Sc * Bc * Ec * 2);       // 4 MB
  half_t* iwh  = (half_t*)allocB((size_t)Jc * Bc * Ec * 2);       // 8 MB
  half_t* Wkvh = (half_t*)allocB((size_t)2 * HIc * Ec * 2);       // 4 MB
  half_t* Wqh  = (half_t*)allocB((size_t)HIc * Ec * 2);           // 2 MB
  half_t* Wph  = (half_t*)allocB((size_t)HIc * Ec * 2);           // 2 MB
  half_t* Woh  = (half_t*)allocB((size_t)Ec * HIc * 2);           // 2 MB
  half_t* rph  = (half_t*)allocB((size_t)Jc * Ec * 2);            // 4 MB
  half_t* quh  = (half_t*)allocB((size_t)Sc * Bc * HIc * 2);      // 4 MB
  half_t* qvh  = (half_t*)allocB((size_t)Sc * Bc * HIc * 2);      // 4 MB
  half_t* kh   = (half_t*)allocB((size_t)Bc * Hc * Jc * Ic * 2);  // 8 MB
  half_t* vh   = (half_t*)allocB((size_t)Bc * Hc * Jc * Ic * 2);  // 8 MB
  half_t* posh = (half_t*)allocB((size_t)Hc * Jc * Ic * 2);       // 4 MB
  half_t* ctxh = (half_t*)allocB((size_t)Sc * Bc * HIc * 2);      // 4 MB
  half_t* Dh   = (half_t*)allocB((size_t)Bc * Hc * Sc * Jc * 2);  // 128 MB
  float*  outf = (float*)allocB((size_t)Sc * Bc * Ec * 4);        // 8 MB

  const int T = 256;
  auto g = [](long n, int d) { return (unsigned)((n + d - 1) / d); };

  k_cvt<<<g((long)Sc * Bc * Ec, T), T, 0, stream>>>(x, xh, Sc * Bc * Ec);
  k_cvt<<<g((long)Jc * Ec, T), T, 0, stream>>>(rel_pos, rph, Jc * Ec);
  k_cvt<<<g((long)2 * HIc * Ec, T), T, 0, stream>>>(Wkv, Wkvh, 2 * HIc * Ec);
  k_cvt<<<g((long)HIc * Ec, T), T, 0, stream>>>(Wq, Wqh, HIc * Ec);
  k_cvt<<<g((long)HIc * Ec, T), T, 0, stream>>>(Wp, Wph, HIc * Ec);
  k_cvt<<<g((long)Ec * HIc, T), T, 0, stream>>>(Wo, Woh, Ec * HIc);
  k_build_iw<<<g((long)Jc * Bc * Ec, T), T, 0, stream>>>(memory, x, iwh, Jc * Bc * Ec);

  // 32x64 tiles per wave
  k_gemm_kv <<<dim3(2 * HIc / 64, Jc * Bc / 32), 32, 0, stream>>>(iwh, Wkvh, kh, vh);
  k_gemm_q  <<<dim3(HIc / 64, Sc * Bc / 32),     32, 0, stream>>>(xh, Wqh, u, v, quh, qvh);
  k_gemm_pos<<<dim3(HIc / 64, Jc / 32),          32, 0, stream>>>(rph, Wph, posh);
  k_gemm_D  <<<dim3(Jc / 64, Sc / 32, Bc * Hc),  32, 0, stream>>>(qvh, posh, Dh);
  k_attn    <<<dim3(Sc / 16, Bc * Hc),           32, 0, stream>>>(quh, kh, vh, Dh, ctxh);
  k_gemm_out<<<dim3(Ec / 64, Sc * Bc / 32),      32, 0, stream>>>(ctxh, Woh, x, outf);
  k_layernorm<<<Sc * Bc, 256, 0, stream>>>(outf, gamma, beta, (float*)d_out);
}
